// TriangleAttentionStartingNode_31756988186830
// MI455X (gfx1250) — matrix operand order repaired and checked by reading, not compiled, not run on hardware
//
#include <hip/hip_runtime.h>

typedef __bf16 bf16;
typedef __attribute__((ext_vector_type(16))) __bf16 v16bf;
typedef __attribute__((ext_vector_type(8)))  float  v8f;

#define SS 256   // sequence
#define CC 128   // channels
#define HH 4     // heads
#define DD 32    // head dim
#define NROW (SS*SS)   // 65536 flattened (i,j) rows

// ---------------------------------------------------------------------------
// WMMA fragment loaders (contiguous k order, same in A and B; wave32)
// A 16x32 (MxK): lane m = lane&15, half = lane>>4; VGPR v holds K = k,k+1 with
//   k = 2v + 16*half
// B 32x16 (KxN): lane n = lane&15, half = lane>>4; VGPR v holds K = k,k+1 with
//   k = 2v + 16*half
// C/D 16x16 f32: VGPR v, lane: row m = v + 8*half, col n = lane&15
// ---------------------------------------------------------------------------

__device__ __forceinline__ v16bf load_fragA(const bf16* p, int rowStride) {
  int lane = threadIdx.x & 31;
  int m = lane & 15, half = lane >> 4;
  const bf16* row = p + (long)m * rowStride;
  union { v16bf v; unsigned u[8]; } r;
#pragma unroll
  for (int q = 0; q < 8; ++q) {
    int k = 2 * q + 16 * half;
    r.u[q] = *(const unsigned*)(row + k);   // bf16 pair, 4B aligned (k even)
  }
  return r.v;
}

// B fragment where source holds B^T row-major: element B[k][n] = p[n*nStride + k]
__device__ __forceinline__ v16bf load_fragBT(const bf16* p, int nStride) {
  int lane = threadIdx.x & 31;
  int n = lane & 15, half = lane >> 4;
  const bf16* row = p + (long)n * nStride;
  union { v16bf v; unsigned u[8]; } r;
#pragma unroll
  for (int q = 0; q < 8; ++q) {
    int k = 2 * q + 16 * half;
    r.u[q] = *(const unsigned*)(row + k);
  }
  return r.v;
}

// B fragment where source holds B row-major: element B[k][n] = p[k*kStride + n]
__device__ __forceinline__ v16bf load_fragB(const bf16* p, int kStride) {
  int lane = threadIdx.x & 31;
  int n = lane & 15, half = lane >> 4;
  union { v16bf v; bf16 e[16]; } r;
#pragma unroll
  for (int q = 0; q < 8; ++q) {
    int k = 2 * q + 16 * half;
    r.e[2 * q]     = p[(long)k * kStride + n];
    r.e[2 * q + 1] = p[(long)(k + 1) * kStride + n];
  }
  return r.v;
}

__device__ __forceinline__ v8f wmma_bf16(v16bf a, v16bf b, v8f c) {
  return __builtin_amdgcn_wmma_f32_16x16x32_bf16(false, a, false, b, (short)0, c,
                                                 false, false);
}

// ---------------------------------------------------------------------------
// fp32 -> bf16 conversion
// ---------------------------------------------------------------------------
__global__ void cvt_kernel(const float* __restrict__ src, bf16* __restrict__ dst, int n) {
  int i = blockIdx.x * blockDim.x + threadIdx.x;
  if (i < n) dst[i] = (bf16)src[i];
}

// ---------------------------------------------------------------------------
// LayerNorm over C=128, one wave (32 lanes x 4 ch) per row, bf16 out
// ---------------------------------------------------------------------------
__global__ __launch_bounds__(256) void ln_kernel(const float* __restrict__ z,
                                                 const float* __restrict__ w,
                                                 const float* __restrict__ b,
                                                 bf16* __restrict__ out) {
  int wave = threadIdx.x >> 5;
  int lane = threadIdx.x & 31;
  long row = (long)blockIdx.x * 8 + wave;
  const float4 v  = ((const float4*)(z + row * CC))[lane];
  float s  = v.x + v.y + v.z + v.w;
  float s2 = v.x * v.x + v.y * v.y + v.z * v.z + v.w * v.w;
#pragma unroll
  for (int m = 1; m < 32; m <<= 1) {
    s  += __shfl_xor(s,  m, 32);
    s2 += __shfl_xor(s2, m, 32);
  }
  float mu  = s * (1.0f / CC);
  float var = s2 * (1.0f / CC) - mu * mu;
  float rs  = rsqrtf(var + 1e-5f);
  const float4 wv = ((const float4*)w)[lane];
  const float4 bv = ((const float4*)b)[lane];
  bf16* o = out + row * CC + lane * 4;
  o[0] = (bf16)((v.x - mu) * rs * wv.x + bv.x);
  o[1] = (bf16)((v.y - mu) * rs * wv.y + bv.y);
  o[2] = (bf16)((v.z - mu) * rs * wv.z + bv.z);
  o[3] = (bf16)((v.w - mu) * rs * wv.w + bv.w);
}

// ---------------------------------------------------------------------------
// Fused projection: in one pass over zln compute
//   q = zln@Wq^T, k = zln@Wk^T, v = zln@Wv^T (bf16),
//   g = sigmoid(zln@Wg^T) (bf16),  bias = zln@Wb^T (fp32, width 4 via padded tile)
// Block: 256 thr = 8 waves, 128 rows. A-fragments loaded once, 132 WMMAs/wave.
// LDS: 4 x 32KB weights + 4KB padded Wb = 132KB.
// ---------------------------------------------------------------------------
__global__ __launch_bounds__(256) void proj_kernel(const bf16* __restrict__ X,
                                                   const bf16* __restrict__ Wq,
                                                   const bf16* __restrict__ Wk,
                                                   const bf16* __restrict__ Wv,
                                                   const bf16* __restrict__ Wg,
                                                   const bf16* __restrict__ Wb,
                                                   bf16* __restrict__ Yq,
                                                   bf16* __restrict__ Yk,
                                                   bf16* __restrict__ Yv,
                                                   bf16* __restrict__ Yg,
                                                   float* __restrict__ Yb) {
  __shared__ bf16 sW[4][CC * CC];   // 128 KB
  __shared__ bf16 sWb[16 * CC];     // 4 KB (rows 4..15 zero)

  {
    const bf16* srcs[4] = {Wq, Wk, Wv, Wg};
#pragma unroll
    for (int w4 = 0; w4 < 4; ++w4) {
      const unsigned* su = (const unsigned*)srcs[w4];
      unsigned* du = (unsigned*)sW[w4];
      for (int i = threadIdx.x; i < CC * CC / 2; i += 256) du[i] = su[i];
    }
    const unsigned* su = (const unsigned*)Wb;
    unsigned* du = (unsigned*)sWb;
    for (int i = threadIdx.x; i < 16 * CC / 2; i += 256)
      du[i] = (i < HH * CC / 2) ? su[i] : 0u;
  }
  __syncthreads();

  const int wave = threadIdx.x >> 5;
  const int lane = threadIdx.x & 31;
  const int n = lane & 15, half = lane >> 4;
  const long rowBase = (long)blockIdx.x * 128 + wave * 16;

  // prefetch next block's A rows (gfx1250 global_prefetch_b8)
  if (blockIdx.x + 1 < NROW / 128)
    __builtin_prefetch(X + (rowBase + 128) * CC, 0, 1);

  v16bf a[4];
#pragma unroll
  for (int kc = 0; kc < 4; ++kc)
    a[kc] = load_fragA(X + rowBase * CC + kc * 32, CC);

  bf16* outs[4] = {Yq, Yk, Yv, Yg};
#pragma unroll
  for (int w4 = 0; w4 < 4; ++w4) {
#pragma unroll
    for (int nt = 0; nt < 8; ++nt) {
      v8f acc = {};
#pragma unroll
      for (int kc = 0; kc < 4; ++kc) {
        v16bf bfrag = load_fragBT(sW[w4] + (nt * 16) * CC + kc * 32, CC);
        acc = wmma_bf16(a[kc], bfrag, acc);
      }
#pragma unroll
      for (int q = 0; q < 8; ++q) {
        int m = q + 8 * half;
        long idx = (rowBase + m) * CC + nt * 16 + n;
        float val = acc[q];
        if (w4 == 3) val = 1.0f / (1.0f + __expf(-val));   // gate: sigmoid
        outs[w4][idx] = (bf16)val;
      }
    }
  }

  // bias tile: 16-wide padded Wb, keep columns n < HH
  {
    v8f acc = {};
#pragma unroll
    for (int kc = 0; kc < 4; ++kc) {
      v16bf bfrag = load_fragBT(sWb + kc * 32, CC);
      acc = wmma_bf16(a[kc], bfrag, acc);
    }
    if (n < HH) {
#pragma unroll
      for (int q = 0; q < 8; ++q) {
        int m = q + 8 * half;
        Yb[(rowBase + m) * HH + n] = acc[q];
      }
    }
  }
}

// ---------------------------------------------------------------------------
// Output GEMM: out(65536x128 fp32) = X(65536x128 bf16) @ W^T (128x128 bf16)
// ---------------------------------------------------------------------------
__global__ __launch_bounds__(256) void gemm_out_kernel(const bf16* __restrict__ X,
                                                       const bf16* __restrict__ W,
                                                       float* __restrict__ Y) {
  __shared__ bf16 sW[CC * CC];  // 32 KB
  {
    const unsigned* Wu = (const unsigned*)W;
    unsigned* sWu = (unsigned*)sW;
    for (int i = threadIdx.x; i < CC * CC / 2; i += 256) sWu[i] = Wu[i];
  }
  __syncthreads();

  const int wave = threadIdx.x >> 5;
  const int lane = threadIdx.x & 31;
  const int n = lane & 15, half = lane >> 4;
  const long rowBase = (long)blockIdx.x * 128 + wave * 16;

  if (blockIdx.x + 1 < NROW / 128)
    __builtin_prefetch(X + (rowBase + 128) * CC, 0, 1);

  v16bf a[4];
#pragma unroll
  for (int kc = 0; kc < 4; ++kc)
    a[kc] = load_fragA(X + rowBase * CC + kc * 32, CC);

#pragma unroll
  for (int nt = 0; nt < 8; ++nt) {
    v8f acc = {};
#pragma unroll
    for (int kc = 0; kc < 4; ++kc) {
      v16bf bfrag = load_fragBT(sW + (nt * 16) * CC + kc * 32, CC);
      acc = wmma_bf16(a[kc], bfrag, acc);
    }
#pragma unroll
    for (int q = 0; q < 8; ++q) {
      int m = q + 8 * half;
      Y[(rowBase + m) * CC + nt * 16 + n] = acc[q];
    }
  }
}

// ---------------------------------------------------------------------------
// Attention: grid (2 i-tiles of 128, j=256, h=4), 256 thr = 8 waves.
// Each wave: 16 query rows. scores (16x256) in 16 WMMA accumulators,
// softmax(+bias), probs->LDS (bf16), P@V (16 WMMAs), gate, bf16 out.
// ---------------------------------------------------------------------------
__global__ __launch_bounds__(256) void attn_kernel(const bf16* __restrict__ Q,
                                                   const bf16* __restrict__ K,
                                                   const bf16* __restrict__ V,
                                                   const float* __restrict__ bias,
                                                   const bf16* __restrict__ G,
                                                   bf16* __restrict__ GO) {
  __shared__ bf16 sK[SS * DD];        // 16 KB
  __shared__ bf16 sV[SS * DD];        // 16 KB
  __shared__ bf16 sP[8 * 16 * SS];    // 64 KB

  const int j  = blockIdx.y;
  const int h  = blockIdx.z;
  const int i0 = blockIdx.x * 128;
  const int tid = threadIdx.x;

  // stage K_{j,h} (256x32) and V_{j,h} into LDS as bf16 pairs
  for (int e = tid; e < SS * DD / 2; e += 256) {
    int kk = (e * 2) >> 5;
    int d  = (e * 2) & 31;
    long src = ((long)(j * SS + kk)) * CC + h * DD + d;
    ((unsigned*)sK)[e] = *(const unsigned*)(K + src);
    ((unsigned*)sV)[e] = *(const unsigned*)(V + src);
  }
  __syncthreads();

  const int wave = tid >> 5, lane = tid & 31;
  const int n = lane & 15, half = lane >> 4;
  const int rowB = i0 + wave * 16;

  // Q fragment: 16 rows (i), K-dim = d (32), row stride = S*C
  v16bf qf = load_fragA(Q + ((long)rowB * SS + j) * CC + h * DD, SS * CC);

  // scores: 16 tiles over k (256)
  v8f acc[16];
#pragma unroll
  for (int t = 0; t < 16; ++t) {
    v16bf kf = load_fragBT(sK + (t * 16) * DD, DD);  // B = K^T tile
    v8f zc = {};
    acc[t] = wmma_bf16(qf, kf, zc);
  }

  // per-row bias (rows m = q + 8*half)
  float bv[8];
#pragma unroll
  for (int q = 0; q < 8; ++q)
    bv[q] = bias[((long)(rowB + q + 8 * half) * SS + j) * HH + h];

  const float scale = 0.17677669529663687f;  // 1/sqrt(32)
  float inv[8];
#pragma unroll
  for (int q = 0; q < 8; ++q) {
    float mx = -1e30f;
#pragma unroll
    for (int t = 0; t < 16; ++t) {
      float s = acc[t][q] * scale + bv[q];
      acc[t][q] = s;
      mx = fmaxf(mx, s);
    }
#pragma unroll
    for (int msk = 1; msk < 16; msk <<= 1) mx = fmaxf(mx, __shfl_xor(mx, msk, 32));
    float sm = 0.f;
#pragma unroll
    for (int t = 0; t < 16; ++t) {
      float e = __expf(acc[t][q] - mx);
      acc[t][q] = e;
      sm += e;
    }
#pragma unroll
    for (int msk = 1; msk < 16; msk <<= 1) sm += __shfl_xor(sm, msk, 32);
    inv[q] = 1.0f / sm;
  }

  // probs -> LDS (D-layout -> row-major bf16), then P @ V
  bf16* myP = sP + wave * 16 * SS;
#pragma unroll
  for (int t = 0; t < 16; ++t)
#pragma unroll
    for (int q = 0; q < 8; ++q) {
      int m = q + 8 * half;
      myP[m * SS + t * 16 + n] = (bf16)(acc[t][q] * inv[q]);
    }
  __syncthreads();

#pragma unroll
  for (int dh = 0; dh < 2; ++dh) {
    v8f o = {};
#pragma unroll
    for (int kc = 0; kc < 8; ++kc) {
      v16bf pa = load_fragA(myP + kc * 32, SS);                 // 16 x 32 probs
      v16bf vbf = load_fragB(sV + (kc * 32) * DD + dh * 16, DD); // 32 x 16 of V
      o = wmma_bf16(pa, vbf, o);
    }
#pragma unroll
    for (int q = 0; q < 8; ++q) {
      int m = q + 8 * half;
      long idx = ((long)(rowB + m) * SS + j) * CC + h * DD + dh * 16 + n;
      float g = (float)G[idx];
      GO[idx] = (bf16)(g * o[q]);
    }
  }
}

// ---------------------------------------------------------------------------
// Host launch
// ---------------------------------------------------------------------------
extern "C" void kernel_launch(void* const* d_in, const int* in_sizes, int n_in,
                              void* d_out, int out_size, void* d_ws, size_t ws_size,
                              hipStream_t stream) {
  const float* z   = (const float*)d_in[0];
  const float* lnw = (const float*)d_in[1];
  const float* lnb = (const float*)d_in[2];
  const float* Wq  = (const float*)d_in[3];
  const float* Wk  = (const float*)d_in[4];
  const float* Wv  = (const float*)d_in[5];
  const float* Wb  = (const float*)d_in[6];
  const float* Wg  = (const float*)d_in[7];
  const float* Wo  = (const float*)d_in[8];

  char* ws = (char*)d_ws;
  const size_t actBytes = (size_t)NROW * CC * sizeof(bf16);
  bf16* zln = (bf16*)ws; ws += actBytes;
  bf16* qb  = (bf16*)ws; ws += actBytes;
  bf16* kb  = (bf16*)ws; ws += actBytes;
  bf16* vb  = (bf16*)ws; ws += actBytes;
  bf16* gb  = (bf16*)ws; ws += actBytes;
  bf16* gob = (bf16*)ws; ws += actBytes;
  bf16* wqb = (bf16*)ws; ws += CC * CC * sizeof(bf16);
  bf16* wkb = (bf16*)ws; ws += CC * CC * sizeof(bf16);
  bf16* wvb = (bf16*)ws; ws += CC * CC * sizeof(bf16);
  bf16* wgb = (bf16*)ws; ws += CC * CC * sizeof(bf16);
  bf16* wob = (bf16*)ws; ws += CC * CC * sizeof(bf16);
  bf16* wbb = (bf16*)ws; ws += HH * CC * sizeof(bf16);
  float* bias = (float*)ws; ws += (size_t)NROW * HH * sizeof(float);

  // weights -> bf16
  cvt_kernel<<<(CC * CC + 255) / 256, 256, 0, stream>>>(Wq, wqb, CC * CC);
  cvt_kernel<<<(CC * CC + 255) / 256, 256, 0, stream>>>(Wk, wkb, CC * CC);
  cvt_kernel<<<(CC * CC + 255) / 256, 256, 0, stream>>>(Wv, wvb, CC * CC);
  cvt_kernel<<<(CC * CC + 255) / 256, 256, 0, stream>>>(Wg, wgb, CC * CC);
  cvt_kernel<<<(CC * CC + 255) / 256, 256, 0, stream>>>(Wo, wob, CC * CC);
  cvt_kernel<<<(HH * CC + 255) / 256, 256, 0, stream>>>(Wb, wbb, HH * CC);

  // layernorm
  ln_kernel<<<NROW / 8, 256, 0, stream>>>(z, lnw, lnb, zln);

  // fused projections (q, k, v, sigmoid-gate, bias)
  proj_kernel<<<NROW / 128, 256, 0, stream>>>(zln, wqb, wkb, wvb, wgb, wbb,
                                              qb, kb, vb, gb, bias);

  // attention + gating
  dim3 ag(2, SS, HH);
  attn_kernel<<<ag, 256, 0, stream>>>(qb, kb, vb, bias, gb, gob);

  // output projection -> fp32
  gemm_out_kernel<<<NROW / 128, 256, 0, stream>>>(gob, wob, (float*)d_out);
}
